// R_GAMLP_RLU_17867063951771
// MI455X (gfx1250) — compile-verified, matrix operand order checked by reading, not run on hardware
//
#include <hip/hip_runtime.h>
#include <hip/hip_bf16.h>

typedef __attribute__((ext_vector_type(16))) _Float16 v16h;
typedef __attribute__((ext_vector_type(8)))  _Float16 v8h;
typedef __attribute__((ext_vector_type(8)))  float    v8f;
typedef int v4i_ __attribute__((vector_size(16)));

#define HOPS 10
#define NN   100000
#define FF   256
#define HID  512
#define CC   47

// GEMM tiling
#define KC    256            // K-chunk staged in LDS
#define LDB   (KC + 8)       // padded row stride (halves) -> conflict-free ds_load_b128
#define BROWS 256            // rows per block (8 waves x 32 rows)
#define BCOLS 64             // cols per block

#if __has_builtin(__builtin_amdgcn_global_load_async_to_lds_b128) && \
    __has_builtin(__builtin_amdgcn_s_wait_asynccnt)
#define USE_ASYNC_LDS 1
#endif

__device__ __forceinline__ float lrelu(float x) { return x >= 0.0f ? x : 0.2f * x; }
__device__ __forceinline__ float prelu(float x, float a) { return x >= 0.0f ? x : a * x; }

// ---------------------------------------------------------------------------
// Kernel 1: per-(hop,node) dual dot products  xl[h,n] = f.wl, xr[h,n] = f.wr
// ---------------------------------------------------------------------------
__global__ __launch_bounds__(256) void hop_dots(const float* __restrict__ feats,
                                                const float* __restrict__ wa,
                                                float* __restrict__ xl,
                                                float* __restrict__ xr) {
  const int wid  = (blockIdx.x * 256 + threadIdx.x) >> 5;   // wave-uniform
  const int lane = threadIdx.x & 31;
  const float4* row = (const float4*)(feats + (size_t)wid * FF);
  const float4* wlv = (const float4*)(wa);
  const float4* wrv = (const float4*)(wa + FF);
  float4 x0 = row[lane], x1 = row[lane + 32];
  float4 l0 = wlv[lane], l1 = wlv[lane + 32];
  float4 r0 = wrv[lane], r1 = wrv[lane + 32];
  float sl = x0.x*l0.x + x0.y*l0.y + x0.z*l0.z + x0.w*l0.w
           + x1.x*l1.x + x1.y*l1.y + x1.z*l1.z + x1.w*l1.w;
  float sr = x0.x*r0.x + x0.y*r0.y + x0.z*r0.z + x0.w*r0.w
           + x1.x*r1.x + x1.y*r1.y + x1.z*r1.z + x1.w*r1.w;
  #pragma unroll
  for (int off = 16; off; off >>= 1) {
    sl += __shfl_xor(sl, off, 32);
    sr += __shfl_xor(sr, off, 32);
  }
  if (lane == 0) { xl[wid] = sl; xr[wid] = sr; }
}

// ---------------------------------------------------------------------------
// Kernel 2: per-node attention recursion -> final softmax weights att[n][10]
// Uses identity history@wl = sum_h att[h] * xl[h].
// ---------------------------------------------------------------------------
__global__ __launch_bounds__(256) void attn_kernel(const float* __restrict__ xl,
                                                   const float* __restrict__ xr,
                                                   const float* __restrict__ ba_p,
                                                   float* __restrict__ att) {
  const int n = blockIdx.x * 256 + threadIdx.x;
  if (n >= NN) return;
  const float ba = ba_p[0];
  float L[HOPS], R[HOPS], s[HOPS];
  #pragma unroll
  for (int h = 0; h < HOPS; ++h) { L[h] = xl[h * NN + n]; R[h] = xr[h * NN + n]; }
  s[0] = lrelu(L[0] + R[0] + ba);
  #pragma unroll
  for (int i = 1; i < HOPS; ++i) {
    float mx = -1e30f;
    for (int h = 0; h < i; ++h) mx = fmaxf(mx, s[h]);
    float den = 0.0f, num = 0.0f;
    for (int h = 0; h < i; ++h) {
      float e = __expf(s[h] - mx);
      den += e; num += e * L[h];
    }
    s[i] = lrelu(num / den + R[i] + ba);
  }
  float mx = -1e30f;
  #pragma unroll
  for (int h = 0; h < HOPS; ++h) mx = fmaxf(mx, s[h]);
  float den = 0.0f, e[HOPS];
  #pragma unroll
  for (int h = 0; h < HOPS; ++h) { e[h] = __expf(s[h] - mx); den += e[h]; }
  const float inv = 1.0f / den;
  #pragma unroll
  for (int h = 0; h < HOPS; ++h) att[(size_t)n * HOPS + h] = e[h] * inv;
}

// ---------------------------------------------------------------------------
// Kernel 3: right[n,f] = sum_h att[n,h] * feats[h,n,f]; emit f16 GEMM operand.
// ---------------------------------------------------------------------------
__global__ __launch_bounds__(256) void weighted_right(const float* __restrict__ feats,
                                                      const float* __restrict__ att,
                                                      _Float16* __restrict__ rightH) {
  const int t = blockIdx.x * 256 + threadIdx.x;      // NN*64 threads
  const int n = t >> 6, q = t & 63;
  float a[HOPS];
  #pragma unroll
  for (int h = 0; h < HOPS; ++h) a[h] = att[(size_t)n * HOPS + h];
  float4 acc = make_float4(0.f, 0.f, 0.f, 0.f);
  #pragma unroll
  for (int h = 0; h < HOPS; ++h) {
    float4 v = ((const float4*)(feats + ((size_t)h * NN + n) * FF))[q];
    acc.x += a[h] * v.x; acc.y += a[h] * v.y; acc.z += a[h] * v.z; acc.w += a[h] * v.w;
  }
  _Float16* o = rightH + (size_t)n * FF + q * 4;
  o[0] = (_Float16)acc.x; o[1] = (_Float16)acc.y; o[2] = (_Float16)acc.z; o[3] = (_Float16)acc.w;
}

// ---------------------------------------------------------------------------
// label_emb (N x 47 f32) -> padded f16 (N x 64), zero-filled tail
// ---------------------------------------------------------------------------
__global__ __launch_bounds__(256) void labpad(const float* __restrict__ lab,
                                              _Float16* __restrict__ labH) {
  const int t = blockIdx.x * 256 + threadIdx.x;      // NN*64 threads
  const int n = t >> 6, c = t & 63;
  labH[t] = (c < CC) ? (_Float16)lab[(size_t)n * CC + c] : (_Float16)0.0f;
}

// ---------------------------------------------------------------------------
// Weight prep: W (K x M, row-major f32) -> Wt (Mpad x Kpad, f16), zero-padded.
// ---------------------------------------------------------------------------
__global__ __launch_bounds__(256) void prep_wt(const float* __restrict__ W,
                                               _Float16* __restrict__ Wt,
                                               int K, int M, int Kpad, int Mpad) {
  const int t = blockIdx.x * 256 + threadIdx.x;
  if (t >= Kpad * Mpad) return;
  const int kp = t % Kpad, mp = t / Kpad;
  float v = (kp < K && mp < M) ? W[(size_t)kp * M + mp] : 0.0f;
  Wt[(size_t)mp * Kpad + kp] = (_Float16)v;
}

// ---------------------------------------------------------------------------
// 16-bit A/B fragment load per CDNA5 ISA layout (16x32, wave32):
// lanes 0-15 hold K = kb+[0..7] and kb+[16..23]; lanes 16-31 hold +8.
// Works for both global (global_load_b128) and LDS (ds_load_b128) pointers.
// ---------------------------------------------------------------------------
__device__ __forceinline__ v16h frag16(const _Float16* p, int ks) {
  const _Float16* q = p + ks;
  v8h lo = *reinterpret_cast<const v8h*>(q);
  v8h hi = *reinterpret_cast<const v8h*>(q + 16);
  v16h r;
  #pragma unroll
  for (int i = 0; i < 8; ++i) { r[i] = lo[i]; r[i + 8] = hi[i]; }
  return r;
}

// ---------------------------------------------------------------------------
// WMMA GEMM: C(N x M) = A(N x K, f16) @ Wt^T (Wt is Mpad x Kpad f16) + epilogue
// Block: 256 threads = 8 waves; block tile 256 rows x 64 cols.
// Wave tile: 32 rows x 64 cols = 8 accumulators; 8 WMMAs per 2 A-frag loads.
// Weight panel staged in LDS (async global->LDS when available).
// EPI 0: +bias -> h0h, xi -> xiH, support -> outH
// EPI 1: +xiH residual -> xi -> xiH, support(0.5 xi + 0.5 h0) -> outH
// EPI 2: +xiH residual -> prelu -> outH
// EPI 3: +bias -> f32 store, col < Mvalid guard (ld = Mvalid)
// EPI 4: +bias -> prelu -> outH
// ---------------------------------------------------------------------------
template <int EPI>
__global__ __launch_bounds__(256) void gemm_wmma(const _Float16* __restrict__ A,
                                                 const _Float16* __restrict__ Wt,
                                                 const float* __restrict__ bias,
                                                 const float* __restrict__ scal,
                                                 float* __restrict__ outF,
                                                 _Float16* __restrict__ outH,
                                                 _Float16* __restrict__ xiH,
                                                 _Float16* __restrict__ h0h,
                                                 int Nrows, int K, int M, int Mvalid) {
  __shared__ _Float16 Bs[BCOLS * LDB];                // 33.8 KB
  const int lane = threadIdx.x & 31;
  const int wv   = threadIdx.x >> 5;
  const int row0 = blockIdx.x * BROWS + wv * 32;      // wave-uniform
  const int col0 = blockIdx.y * BCOLS;
  const int lr = lane & 15;
  const int ks = (lane & 16) ? 8 : 0;

  v8f acc[2][4] = {};
  // clamp A row indices so partial last block stays in-bounds (stores guarded)
  const int rA0 = (row0 + lr      < Nrows) ? (row0 + lr)      : (Nrows - 1);
  const int rA1 = (row0 + 16 + lr < Nrows) ? (row0 + 16 + lr) : (Nrows - 1);
  const _Float16* arow0 = A + (size_t)rA0 * K;
  const _Float16* arow1 = A + (size_t)rA1 * K;

  for (int kc0 = 0; kc0 < K; kc0 += KC) {
    const int kcLen = (K - kc0 < KC) ? (K - kc0) : KC;
    const int k8n   = kcLen >> 3;                     // 16B chunks per column
    __syncthreads();
    for (int c = threadIdx.x; c < BCOLS * k8n; c += 256) {
      const int col = c / k8n;
      const int k8  = c - col * k8n;
      const _Float16* src = Wt + (size_t)(col0 + col) * K + kc0 + k8 * 8;
      _Float16* dst = &Bs[col * LDB + k8 * 8];
#ifdef USE_ASYNC_LDS
      __builtin_amdgcn_global_load_async_to_lds_b128(
          (__attribute__((address_space(1))) v4i_*)(uintptr_t)src,
          (__attribute__((address_space(3))) v4i_*)(unsigned int)(uintptr_t)dst,
          0, 0);
#else
      *reinterpret_cast<v8h*>(dst) = *reinterpret_cast<const v8h*>(src);
#endif
    }
#ifdef USE_ASYNC_LDS
    __builtin_amdgcn_s_wait_asynccnt(0);
#endif
    __syncthreads();

    for (int kb = 0; kb < kcLen; kb += 32) {
      v16h a0 = frag16(arow0 + kc0 + kb, ks);
      v16h a1 = frag16(arow1 + kc0 + kb, ks);
      #pragma unroll
      for (int t = 0; t < 4; ++t) {
        v16h bf = frag16(&Bs[(t * 16 + lr) * LDB + kb], ks);
        acc[0][t] = __builtin_amdgcn_wmma_f32_16x16x32_f16(
            false, a0, false, bf, (short)0, acc[0][t], false, false);
        acc[1][t] = __builtin_amdgcn_wmma_f32_16x16x32_f16(
            false, a1, false, bf, (short)0, acc[1][t], false, false);
      }
    }
  }

  const int cRowOff = (lane >> 4) << 3;               // 0 or 8
  const float alpha = scal ? scal[0] : 0.0f;
  #pragma unroll
  for (int g = 0; g < 2; ++g) {
    #pragma unroll
    for (int t = 0; t < 4; ++t) {
      const int col = col0 + t * 16 + lr;
      float bcol = 0.0f;
      if constexpr (EPI == 0 || EPI == 3 || EPI == 4) bcol = bias[col];
      #pragma unroll
      for (int r = 0; r < 8; ++r) {
        const int row = row0 + g * 16 + cRowOff + r;
        if (row >= Nrows) continue;
        const size_t idx = (size_t)row * M + col;
        float v = acc[g][t][r];
        if constexpr (EPI == 0) {
          v += bcol;
          h0h[idx] = (_Float16)v;
          float xi = prelu(v, alpha);
          xiH[idx] = (_Float16)xi;
          outH[idx] = (_Float16)(0.5f * (xi + v));    // ALPHA = 0.5
        } else if constexpr (EPI == 1) {
          v += (float)xiH[idx];
          float xi = prelu(v, alpha);
          xiH[idx] = (_Float16)xi;
          outH[idx] = (_Float16)(0.5f * xi + 0.5f * (float)h0h[idx]);
        } else if constexpr (EPI == 2) {
          v += (float)xiH[idx];
          outH[idx] = (_Float16)prelu(v, alpha);
        } else if constexpr (EPI == 3) {
          if (col < Mvalid) outF[(size_t)row * Mvalid + col] = v + bcol;
        } else {                                      // EPI == 4
          outH[idx] = (_Float16)prelu(v + bcol, alpha);
        }
      }
    }
  }
}

// ---------------------------------------------------------------------------
// Final merge: out = feature-branch + label-branch (both N x 47 f32)
// ---------------------------------------------------------------------------
__global__ __launch_bounds__(256) void add_out(const float* __restrict__ xp,
                                               const float* __restrict__ yp,
                                               float* __restrict__ out) {
  const int t = blockIdx.x * 256 + threadIdx.x;
  if (t >= NN * CC) return;
  out[t] = xp[t] + yp[t];
}

// ---------------------------------------------------------------------------

extern "C" void kernel_launch(void* const* d_in, const int* in_sizes, int n_in,
                              void* d_out, int out_size, void* d_ws, size_t ws_size,
                              hipStream_t stream) {
  const float* feats   = (const float*)d_in[0];
  const float* labemb  = (const float*)d_in[1];
  const float* wa      = (const float*)d_in[2];
  const float* ba      = (const float*)d_in[3];
  const float* w0      = (const float*)d_in[4];
  const float* b0      = (const float*)d_in[5];
  const float* wg1     = (const float*)d_in[6];
  const float* wg2     = (const float*)d_in[7];
  const float* w_last  = (const float*)d_in[8];
  const float* b_last  = (const float*)d_in[9];
  const float* a_out   = (const float*)d_in[10];
  const float* wl0     = (const float*)d_in[11];
  const float* bl0     = (const float*)d_in[12];
  const float* wl1     = (const float*)d_in[13];
  const float* bl1     = (const float*)d_in[14];
  const float* wl2     = (const float*)d_in[15];
  const float* bl2     = (const float*)d_in[16];
  const float* wl3     = (const float*)d_in[17];
  const float* bl3     = (const float*)d_in[18];
  const float* a_lab   = (const float*)d_in[19];
  float* out = (float*)d_out;

  // ---- workspace carve-out (deterministic) ----
  char* wsp = (char*)d_ws;
  auto carve = [&](size_t bytes) -> void* {
    void* p = wsp;
    wsp += (bytes + 255) & ~(size_t)255;
    return p;
  };
  float*    xl     = (float*)   carve((size_t)HOPS * NN * 4);
  float*    xr     = (float*)   carve((size_t)HOPS * NN * 4);
  float*    att    = (float*)   carve((size_t)NN * HOPS * 4);
  _Float16* rightH = (_Float16*)carve((size_t)NN * FF * 2);
  _Float16* h0h    = (_Float16*)carve((size_t)NN * HID * 2);
  _Float16* xiH    = (_Float16*)carve((size_t)NN * HID * 2);
  _Float16* sh0    = (_Float16*)carve((size_t)NN * HID * 2);
  _Float16* sh1    = (_Float16*)carve((size_t)NN * HID * 2);
  _Float16* labH   = (_Float16*)carve((size_t)NN * 64 * 2);
  float*    xpart  = (float*)   carve((size_t)NN * CC * 4);
  float*    ypart  = (float*)   carve((size_t)NN * CC * 4);
  _Float16* w0t    = (_Float16*)carve((size_t)HID * FF * 2);     // [512][256]
  _Float16* wg1t   = (_Float16*)carve((size_t)HID * HID * 2);    // [512][512]
  _Float16* wg2t   = (_Float16*)carve((size_t)HID * HID * 2);
  _Float16* wlastt = (_Float16*)carve((size_t)64 * HID * 2);     // [64][512]
  _Float16* wl0t   = (_Float16*)carve((size_t)HID * 64 * 2);     // [512][64]
  _Float16* wl1t   = (_Float16*)carve((size_t)HID * HID * 2);
  _Float16* wl2t   = (_Float16*)carve((size_t)HID * HID * 2);
  _Float16* wl3t   = (_Float16*)carve((size_t)64 * HID * 2);     // [64][512]

  auto blocks = [](size_t n) { return (int)((n + 255) / 256); };

  // ---- weight prep (transpose + f16 + pad) ----
  prep_wt<<<blocks((size_t)HID * FF),  256, 0, stream>>>(w0,     w0t,    FF,  HID, FF,  HID);
  prep_wt<<<blocks((size_t)HID * HID), 256, 0, stream>>>(wg1,    wg1t,   HID, HID, HID, HID);
  prep_wt<<<blocks((size_t)HID * HID), 256, 0, stream>>>(wg2,    wg2t,   HID, HID, HID, HID);
  prep_wt<<<blocks((size_t)64 * HID),  256, 0, stream>>>(w_last, wlastt, HID, CC,  HID, 64);
  prep_wt<<<blocks((size_t)HID * 64),  256, 0, stream>>>(wl0,    wl0t,   CC,  HID, 64,  HID);
  prep_wt<<<blocks((size_t)HID * HID), 256, 0, stream>>>(wl1,    wl1t,   HID, HID, HID, HID);
  prep_wt<<<blocks((size_t)HID * HID), 256, 0, stream>>>(wl2,    wl2t,   HID, HID, HID, HID);
  prep_wt<<<blocks((size_t)64 * HID),  256, 0, stream>>>(wl3,    wl3t,   HID, CC,  HID, 64);

  // ---- attention path ----
  hop_dots<<<(HOPS * NN) / 8, 256, 0, stream>>>(feats, wa, xl, xr);
  attn_kernel<<<blocks(NN), 256, 0, stream>>>(xl, xr, ba, att);
  weighted_right<<<(NN * 64) / 256, 256, 0, stream>>>(feats, att, rightH);
  labpad<<<(NN * 64) / 256, 256, 0, stream>>>(labemb, labH);

  const int gx = (NN + BROWS - 1) / BROWS;             // 391
  dim3 blk(256);

  // ---- main branch GEMMs ----
  gemm_wmma<0><<<dim3(gx, HID / 64), blk, 0, stream>>>(rightH, w0t, b0, a_out,
      nullptr, sh0, xiH, h0h, NN, FF, HID, HID);
  gemm_wmma<1><<<dim3(gx, HID / 64), blk, 0, stream>>>(sh0, wg1t, nullptr, a_out,
      nullptr, sh1, xiH, h0h, NN, HID, HID, HID);
  gemm_wmma<2><<<dim3(gx, HID / 64), blk, 0, stream>>>(sh1, wg2t, nullptr, a_out,
      nullptr, sh0, xiH, h0h, NN, HID, HID, HID);
  gemm_wmma<3><<<dim3(gx, 1), blk, 0, stream>>>(sh0, wlastt, b_last, nullptr,
      xpart, nullptr, nullptr, nullptr, NN, HID, 64, CC);

  // ---- label branch GEMMs (reuse sh0/sh1 ping-pong after main branch) ----
  gemm_wmma<4><<<dim3(gx, HID / 64), blk, 0, stream>>>(labH, wl0t, bl0, a_lab,
      nullptr, sh1, nullptr, nullptr, NN, 64, HID, HID);
  gemm_wmma<4><<<dim3(gx, HID / 64), blk, 0, stream>>>(sh1, wl1t, bl1, a_lab,
      nullptr, sh0, nullptr, nullptr, NN, HID, HID, HID);
  gemm_wmma<4><<<dim3(gx, HID / 64), blk, 0, stream>>>(sh0, wl2t, bl2, a_lab,
      nullptr, sh1, nullptr, nullptr, NN, HID, HID, HID);
  gemm_wmma<3><<<dim3(gx, 1), blk, 0, stream>>>(sh1, wl3t, bl3, nullptr,
      ypart, nullptr, nullptr, nullptr, NN, HID, 64, CC);

  // ---- merge ----
  add_out<<<blocks((size_t)NN * CC), 256, 0, stream>>>(xpart, ypart, out);
}
